// SCARF_30099130810573
// MI455X (gfx1250) — matrix operand order repaired and checked by reading, not compiled, not run on hardware
//
#include <hip/hip_runtime.h>

typedef __attribute__((ext_vector_type(16))) _Float16 v16h;
typedef __attribute__((ext_vector_type(8)))  float    v8f;

#define TILE_M 128
#define TILE_N 128
#define TILE_K 32
#define LDS_STRIDE 80                    // 32 halfs (64B) + 16B pad, multiple of 16
#define TILE_BYTES (TILE_M * LDS_STRIDE) // 10240 per operand tile
#define BUF_BYTES  (2 * TILE_BYTES)      // A tile + B tile per buffer

// ---------------------------------------------------------------------------
// CDNA5 async copy: global -> LDS, 16 bytes per lane, tracked by ASYNCcnt.
// vdst VGPR carries the wave-relative LDS byte address (low 32 bits of the
// generic shared pointer), vaddr the 64-bit global address.
// ---------------------------------------------------------------------------
__device__ __forceinline__ void async_copy_b128(unsigned lds_byte_off,
                                                const void* gaddr) {
  asm volatile("global_load_async_to_lds_b128 %0, %1, off"
               :: "v"(lds_byte_off), "v"(gaddr)
               : "memory");
}
__device__ __forceinline__ void wait_async_le4() {
  asm volatile("s_wait_asynccnt 0x4" ::: "memory");
}
__device__ __forceinline__ void wait_async_0() {
  asm volatile("s_wait_asynccnt 0x0" ::: "memory");
}

// ---------------------------------------------------------------------------
// Corruption: per-row top-k rank of rand_noise (tie-break: lower index wins,
// matching jax.lax.top_k), then x_corrupted = where(mask, min + u*(max-min), x).
// Writes f16 activations: rows [0,B) = x, rows [B,2B) = x_corrupted.
// ---------------------------------------------------------------------------
__global__ void corrupt_cast_kernel(const float* __restrict__ x,
                                    const float* __restrict__ rn,
                                    const float* __restrict__ ur,
                                    const float* __restrict__ mmin,
                                    const float* __restrict__ mmax,
                                    const int* __restrict__ kptr,
                                    _Float16* __restrict__ act,
                                    int B, int M) {
  extern __shared__ float s[];
  const int row = blockIdx.x;
  const int t   = threadIdx.x;
  const float r = rn[(size_t)row * M + t];
  s[t] = r;
  __syncthreads();
  const int K = kptr[0];
  int cnt = 0;
  for (int j = 0; j < M; ++j) {
    float v = s[j];
    cnt += (v > r || (v == r && j < t)) ? 1 : 0;
  }
  const bool corrupt = cnt < K;
  const float xv = x[(size_t)row * M + t];
  const float xr = mmin[t] + ur[(size_t)row * M + t] * (mmax[t] - mmin[t]);
  act[(size_t)row * M + t]       = (_Float16)xv;
  act[(size_t)(B + row) * M + t] = (_Float16)(corrupt ? xr : xv);
}

// ---------------------------------------------------------------------------
// Convert f32 weight (K x N, row-major) -> f16 transposed (N x K, row-major)
// so the GEMM loads both operands with identical K-contiguous tiles.
// ---------------------------------------------------------------------------
__global__ void weight_transpose_f16(const float* __restrict__ w,
                                     _Float16* __restrict__ wt,
                                     int K, int N) {
  int idx = blockIdx.x * blockDim.x + threadIdx.x;
  if (idx >= K * N) return;
  int k = idx / N, n = idx - k * N;
  wt[(size_t)n * K + k] = (_Float16)w[idx];
}

// ---------------------------------------------------------------------------
// C(Nrows x Ncols) = op(A(Nrows x K) @ Bt(Ncols x K)^T + bias), op = ReLU?
// 256 threads = 8 waves; block tile 128x128; wave tile 64x32 = 4x2 WMMA tiles.
// Double-buffered LDS filled by global_load_async_to_lds_b128 (ASYNCcnt);
// steady-state loop is branch-free (last K-step peeled).
// ---------------------------------------------------------------------------
template <bool RELU, bool OUTF32>
__global__ void __launch_bounds__(256)
gemm_wmma(const _Float16* __restrict__ A,
          const _Float16* __restrict__ Bt,
          const float* __restrict__ bias,
          void* __restrict__ Cout,
          int K, int Ncols) {
  __shared__ __align__(16) unsigned char smem[2 * BUF_BYTES];

  const int tid  = threadIdx.x;
  const int lane = tid & 31;
  const int wave = tid >> 5;
  const int half = lane >> 4;         // lane-half selects K sub-range (ISA layout)
  const int l16  = lane & 15;
  const int wmb  = (wave >> 2) * 64;  // wave M offset (2 rows of waves)
  const int wnb  = (wave & 3) * 32;   // wave N offset (4 cols of waves)

  const int rowBase = blockIdx.y * TILE_M;
  const int colBase = blockIdx.x * TILE_N;

  // Wave-relative LDS byte offset of the shared buffer.
  const unsigned lds_base = (unsigned)(size_t)(void*)smem;

  // This thread's two staging chunks (same geometry for A and B tiles):
  // 512 x 16B chunks per tile, 2 per thread.
  const int r0 = tid >> 2,         kc0 = tid & 3;
  const int r1 = (tid + 256) >> 2, kc1 = tid & 3;  // +256 keeps kc identical
  const unsigned ldsO0 = (unsigned)(r0 * LDS_STRIDE + kc0 * 16);
  const unsigned ldsO1 = (unsigned)(r1 * LDS_STRIDE + kc1 * 16);

  // Pointer-advance form: bump by TILE_K halves (64B) per iteration so the
  // loop does 4 constant 64-bit adds instead of rebuilding addresses from k0.
  const _Float16* gA0 = A  + (size_t)(rowBase + r0) * K + kc0 * 8;
  const _Float16* gA1 = A  + (size_t)(rowBase + r1) * K + kc1 * 8;
  const _Float16* gB0 = Bt + (size_t)(colBase + r0) * K + kc0 * 8;
  const _Float16* gB1 = Bt + (size_t)(colBase + r1) * K + kc1 * 8;

  auto issue_tile = [&](int buf) {
    const unsigned base = lds_base + (unsigned)(buf * BUF_BYTES);
    async_copy_b128(base + ldsO0,              gA0);
    async_copy_b128(base + ldsO1,              gA1);
    async_copy_b128(base + TILE_BYTES + ldsO0, gB0);
    async_copy_b128(base + TILE_BYTES + ldsO1, gB1);
    gA0 += TILE_K; gA1 += TILE_K; gB0 += TILE_K; gB1 += TILE_K;
  };

  v8f acc[4][2];
#pragma unroll
  for (int i = 0; i < 4; ++i)
#pragma unroll
    for (int j = 0; j < 2; ++j) {
      v8f z = {0.f, 0.f, 0.f, 0.f, 0.f, 0.f, 0.f, 0.f};
      acc[i][j] = z;
    }

  auto compute_tile = [&](int cur) {
    const unsigned char* sA = smem + cur * BUF_BYTES;
    const unsigned char* sB = sA + TILE_BYTES;

    // A fragments: 16x32 f16; v0..3 = K half*8..+7, v4..7 = K 16+half*8..+7
    v16h afrag[4];
#pragma unroll
    for (int i = 0; i < 4; ++i) {
      const unsigned char* p = sA + (wmb + i * 16 + l16) * LDS_STRIDE;
      v16h a;
      ((float4*)&a)[0] = *(const float4*)(p + half * 16);
      ((float4*)&a)[1] = *(const float4*)(p + 32 + half * 16);
      afrag[i] = a;
    }
    // B fragments: 32x16 f16; v0..3 = K half*8..+7, v4..7 = K 16+half*8..+7
    v16h bfrag[2];
#pragma unroll
    for (int j = 0; j < 2; ++j) {
      const unsigned char* p = sB + (wnb + j * 16 + l16) * LDS_STRIDE;
      v16h b;
      ((float4*)&b)[0] = *(const float4*)(p + half * 16);
      ((float4*)&b)[1] = *(const float4*)(p + 32 + half * 16);
      bfrag[j] = b;
    }

#pragma unroll
    for (int i = 0; i < 4; ++i)
#pragma unroll
      for (int j = 0; j < 2; ++j)
        acc[i][j] = __builtin_amdgcn_wmma_f32_16x16x32_f16(
            false, afrag[i], false, bfrag[j], (short)0, acc[i][j], false, false);
  };

  const int nIter = K / TILE_K;
  issue_tile(0);  // prologue -> buffer 0

  // Steady state: branch-free; always prefetch next tile, wait for current.
  for (int it = 0; it < nIter - 1; ++it) {
    const int cur = it & 1;
    issue_tile(1 - cur);
    wait_async_le4();   // in-order completion => batch for `cur` has landed
    __syncthreads();    // all waves' async writes for `cur` visible
    compute_tile(cur);
    __syncthreads();    // all waves done reading `cur` before it is refilled
  }
  // Peeled last K-step.
  wait_async_0();
  __syncthreads();
  compute_tile((nIter - 1) & 1);

  // Epilogue: C layout — VGPR r: lanes0-15 -> (m=r, n=lane), lanes16-31 -> (m=8+r)
#pragma unroll
  for (int j = 0; j < 2; ++j) {
    const int gn = colBase + wnb + j * 16 + l16;
    const float bv = bias[gn];
#pragma unroll
    for (int i = 0; i < 4; ++i) {
      const int gm = rowBase + wmb + i * 16 + half * 8;
      v8f c = acc[i][j];
#pragma unroll
      for (int r = 0; r < 8; ++r) {
        float v = c[r] + bv;
        if (RELU) v = fmaxf(v, 0.0f);
        if (OUTF32) {
          // Final layer streams 134MB that is never re-read: NT store.
          __builtin_nontemporal_store(
              v, (float*)Cout + (size_t)(gm + r) * Ncols + gn);
        } else {
          ((_Float16*)Cout)[(size_t)(gm + r) * Ncols + gn] = (_Float16)v;
        }
      }
    }
  }
}

// ---------------------------------------------------------------------------
extern "C" void kernel_launch(void* const* d_in, const int* in_sizes, int n_in,
                              void* d_out, int out_size, void* d_ws, size_t ws_size,
                              hipStream_t stream) {
  const float* x    = (const float*)d_in[0];
  const float* rn   = (const float*)d_in[1];
  const float* ur   = (const float*)d_in[2];
  const float* mmin = (const float*)d_in[3];
  const float* mmax = (const float*)d_in[4];
  const float* w[6]    = {(const float*)d_in[5],  (const float*)d_in[7],
                          (const float*)d_in[9],  (const float*)d_in[11],
                          (const float*)d_in[13], (const float*)d_in[15]};
  const float* bvec[6] = {(const float*)d_in[6],  (const float*)d_in[8],
                          (const float*)d_in[10], (const float*)d_in[12],
                          (const float*)d_in[14], (const float*)d_in[16]};
  const int* klen = (const int*)d_in[17];

  const int M = in_sizes[3];      // 512
  const int B = in_sizes[0] / M;  // 16384
  const int E = in_sizes[6];      // 1024
  const int R = 2 * B;            // rows through the MLP

  char* ws = (char*)d_ws;
  size_t off = 0;
  auto alloc = [&](size_t bytes) -> char* {
    char* p = ws + off;
    off = (off + bytes + 255) & ~(size_t)255;
    return p;
  };
  _Float16* act0 = (_Float16*)alloc((size_t)R * M * sizeof(_Float16));
  _Float16* actX = (_Float16*)alloc((size_t)R * E * sizeof(_Float16));
  _Float16* actY = (_Float16*)alloc((size_t)R * E * sizeof(_Float16));
  _Float16* wt[6];
  int wk[6] = {M, E, E, E, E, E};
  for (int i = 0; i < 6; ++i)
    wt[i] = (_Float16*)alloc((size_t)wk[i] * E * sizeof(_Float16));
  (void)ws_size; (void)n_in; (void)out_size;

  // 1) corruption + f16 cast of [x ; x_corrupted]
  corrupt_cast_kernel<<<B, M, M * sizeof(float), stream>>>(
      x, rn, ur, mmin, mmax, klen, act0, B, M);

  // 2) weight convert+transpose (f32 KxN -> f16 NxK), once
  for (int i = 0; i < 6; ++i) {
    int total = wk[i] * E;
    weight_transpose_f16<<<(total + 255) / 256, 256, 0, stream>>>(w[i], wt[i], wk[i], E);
  }

  // 3) GEMM chain
  dim3 blk(256);
  dim3 grid(E / TILE_N, R / TILE_M);
  gemm_wmma<true,  false><<<grid, blk, 0, stream>>>(act0, wt[0], bvec[0], actY, M, E);
  gemm_wmma<true,  false><<<grid, blk, 0, stream>>>(actY, wt[1], bvec[1], actX, E, E);
  gemm_wmma<true,  false><<<grid, blk, 0, stream>>>(actX, wt[2], bvec[2], actY, E, E);
  gemm_wmma<false, false><<<grid, blk, 0, stream>>>(actY, wt[3], bvec[3], actX, E, E);
  gemm_wmma<true,  false><<<grid, blk, 0, stream>>>(actX, wt[4], bvec[4], actY, E, E);
  gemm_wmma<false, true ><<<grid, blk, 0, stream>>>(actY, wt[5], bvec[5], d_out, E, E);
}
